// SlotAttention_87368224735628
// MI455X (gfx1250) — matrix operand order repaired and checked by reading, not compiled, not run on hardware
//
#include <hip/hip_runtime.h>
#include <hip/hip_bf16.h>
#include <math.h>

// ---------------------------------------------------------------------------
// Slot Attention (B=64, NS=11, NK=4096, C=128, 2 iters) for gfx1250 (CDNA5).
// bf16 WMMA (v_wmma_f32_16x16x32_bf16) everywhere; K/V cached in bf16 in ws,
// V stored transposed so a@v B-fragments are contiguous; single pass over
// keys per iteration via U/S accumulation (renormalize-after trick).
// Workgroup-level LDS reduction + per-chunk partials instead of global
// atomics; global_load_async_to_lds staging of the K stream (ASYNCcnt path).
// ---------------------------------------------------------------------------

typedef __attribute__((ext_vector_type(16))) __bf16 v16bf;
typedef __attribute__((ext_vector_type(8)))  float  v8f;
typedef __attribute__((ext_vector_type(4)))  int    v4i;

union F8 { v8f v; float f[8]; };

#define LN_EPS_   1e-5f
#define ATTN_EPS_ 1e-8f

constexpr int B_  = 64;
constexpr int NS_ = 11;
constexpr int NK_ = 4096;
constexpr int C_  = 128;

#define XSTR 136   // padded row stride (halfs) for 128-wide LDS tiles
#define WSTR 136
#define QSTR 136
#define ASTR 40    // padded row stride for 16x32 attention-weight tiles

#if defined(__HIP_DEVICE_COMPILE__) && \
    __has_builtin(__builtin_amdgcn_global_load_async_to_lds_b128) && \
    __has_builtin(__builtin_amdgcn_s_wait_asynccnt)
#define USE_ASYNC_LDS 1
#else
#define USE_ASYNC_LDS 0
#endif

#if USE_ASYNC_LDS
__device__ __forceinline__ void async_ld_b128(const __bf16* gsrc, __bf16* ldst) {
  __builtin_amdgcn_global_load_async_to_lds_b128(
      (__attribute__((address_space(1))) v4i*)gsrc,
      (__attribute__((address_space(3))) v4i*)ldst, 0, 0);
}
#endif

__device__ __forceinline__ v8f wmma_bf16(v16bf a, v16bf b, v8f c) {
  // D = A(16x32) * B(32x16) + C, f32 accum
  return __builtin_amdgcn_wmma_f32_16x16x32_bf16(
      false, a, false, b, (short)0, c, false, false);
}

// A fragment (16x32 bf16, row-major source, padded stride in halfs).
// lane<16: m=lane,  K-chunks at [kb+0..7]  and [kb+16..23]
// lane>=16: m=lane-16, K-chunks at [kb+8..15] and [kb+24..31]
__device__ __forceinline__ v16bf load_frag_a(const __bf16* base, int stride, int kb) {
  int lane = threadIdx.x & 31;
  const __bf16* p = base + (size_t)(lane & 15) * stride + kb + ((lane >> 4) << 3);
  v16bf r;
  float4* rp = reinterpret_cast<float4*>(&r);
  rp[0] = *reinterpret_cast<const float4*>(p);
  rp[1] = *reinterpret_cast<const float4*>(p + 16);
  return r;
}

// B fragment (32x16 bf16) from a transposed source: row n holds K contiguous.
// lane<16: n=lane, K=[kb+0..7] and [kb+16..23]; lane>=16: n=lane-16, K=[kb+8..15] and [kb+24..31]
__device__ __forceinline__ v16bf load_frag_bt(const __bf16* baseT, size_t stride, int kb) {
  int lane = threadIdx.x & 31;
  const __bf16* p = baseT + (size_t)(lane & 15) * stride + kb + ((lane >> 4) << 3);
  v16bf r;
  float4* rp = reinterpret_cast<float4*>(&r);
  rp[0] = *reinterpret_cast<const float4*>(p);
  rp[1] = *reinterpret_cast<const float4*>(p + 16);
  return r;
}

// D fragment: lane<16 -> col n=lane, rows m=j; lane>=16 -> col n=lane-16, rows m=8+j
__device__ __forceinline__ void store8_bf16(const F8& d, __bf16* p) {
  union { __bf16 h[8]; float4 f4; } u;
#pragma unroll
  for (int j = 0; j < 8; ++j) u.h[j] = (__bf16)d.f[j];
  *reinterpret_cast<float4*>(p) = u.f4;
}

__device__ __forceinline__ float wave_sum32(float x) {
#pragma unroll
  for (int m = 16; m >= 1; m >>= 1) x += __shfl_xor(x, m, 32);
  return x;
}

__device__ __forceinline__ F8 f8_zero() {
  F8 r;
#pragma unroll
  for (int j = 0; j < 8; ++j) r.f[j] = 0.f;
  return r;
}

// ---------------------------------------------------------------------------
// K1: fused LayerNorm(inputs) + K/V projection.  Writes k[b][key][c] bf16 and
// vT[b][c][key] bf16.  Grid: B * NK/64 blocks of 256 (8 waves, 64-key tile).
// ---------------------------------------------------------------------------
__global__ __launch_bounds__(256) void k_lnkv(
    const float* __restrict__ xin, const float* __restrict__ g,
    const float* __restrict__ bparm, const float* __restrict__ Wk,
    const float* __restrict__ Wv, __bf16* __restrict__ k_ws,
    __bf16* __restrict__ vT_ws)
{
  __shared__ __bf16 xb[64 * XSTR];     // LN'd input tile, bf16
  __shared__ __bf16 wt[128 * WSTR];    // transposed weight (phased: Wk then Wv)

  const int tid = threadIdx.x, lane = tid & 31, wave = tid >> 5;
  const int b    = blockIdx.x >> 6;          // NK/64 = 64 key tiles
  const int key0 = (blockIdx.x & 63) * 64;

  // --- LayerNorm 64 rows (8 rows per wave), write bf16 to LDS -------------
  float4 gv = reinterpret_cast<const float4*>(g)[lane];
  float4 bv = reinterpret_cast<const float4*>(bparm)[lane];
  const float* xrow = xin + ((size_t)b * NK_ + key0) * C_;
#pragma unroll
  for (int i = 0; i < 8; ++i) {
    int r = wave * 8 + i;
    float4 x = reinterpret_cast<const float4*>(xrow + (size_t)r * C_)[lane];
    float mu = wave_sum32(x.x + x.y + x.z + x.w) * (1.f / 128.f);
    float d0 = x.x - mu, d1 = x.y - mu, d2 = x.z - mu, d3 = x.w - mu;
    float var = wave_sum32(d0*d0 + d1*d1 + d2*d2 + d3*d3) * (1.f / 128.f);
    float rs = rsqrtf(var + LN_EPS_);
    __bf16* dst = &xb[r * XSTR + lane * 4];
    dst[0] = (__bf16)(d0 * rs * gv.x + bv.x);
    dst[1] = (__bf16)(d1 * rs * gv.y + bv.y);
    dst[2] = (__bf16)(d2 * rs * gv.z + bv.z);
    dst[3] = (__bf16)(d3 * rs * gv.w + bv.w);
  }

  const int n = lane & 15, half = (lane >> 4) << 3;

  // --- Phase K: D'[o,key] = Wk^T * xb^T  -> store k[key][o] contiguously ---
  __syncthreads();
  for (int idx = tid; idx < 128 * 128; idx += 256) {
    int i = idx >> 7, o = idx & 127;
    wt[o * WSTR + i] = (__bf16)Wk[idx];
  }
  __syncthreads();
  for (int t = wave; t < 32; t += 8) {   // 8 o-tiles x 4 key-tiles
    int ot = t >> 2, kt = t & 3;
    F8 acc = f8_zero();
#pragma unroll
    for (int kb = 0; kb < 128; kb += 32)
      acc.v = wmma_bf16(load_frag_a(&wt[ot * 16 * WSTR], WSTR, kb),
                        load_frag_bt(&xb[kt * 16 * XSTR], XSTR, kb), acc.v);
    int key = key0 + kt * 16 + n;
    store8_bf16(acc, k_ws + ((size_t)b * NK_ + key) * C_ + ot * 16 + half);
  }

  // --- Phase V: D[key,o] = xb * Wv  -> store vT[o][key] contiguously -------
  __syncthreads();
  for (int idx = tid; idx < 128 * 128; idx += 256) {
    int i = idx >> 7, o = idx & 127;
    wt[o * WSTR + i] = (__bf16)Wv[idx];
  }
  __syncthreads();
  for (int t = wave; t < 32; t += 8) {   // 4 key-tiles x 8 o-tiles
    int kt = t >> 3, ot = t & 7;
    F8 acc = f8_zero();
#pragma unroll
    for (int kb = 0; kb < 128; kb += 32)
      acc.v = wmma_bf16(load_frag_a(&xb[kt * 16 * XSTR], XSTR, kb),
                        load_frag_bt(&wt[ot * 16 * WSTR], WSTR, kb), acc.v);
    int chan = ot * 16 + n;
    store8_bf16(acc, vT_ws + ((size_t)b * C_ + chan) * NK_ + key0 + kt * 16 + half);
  }
}

// ---------------------------------------------------------------------------
// K2: q = LN(query) @ Wq * scale -> qnp[b][16][128] bf16 (slots padded to 16,
// pad rows zero). Grid: B blocks of 128 (4 waves).
// ---------------------------------------------------------------------------
__global__ __launch_bounds__(128) void k_qn(
    const float* __restrict__ qsrc, const float* __restrict__ g,
    const float* __restrict__ bp, const float* __restrict__ Wq,
    __bf16* __restrict__ qnp)
{
  __shared__ __bf16 ql[16 * QSTR];
  __shared__ __bf16 wt[128 * WSTR];
  const int tid = threadIdx.x, lane = tid & 31, wave = tid >> 5;
  const int b = blockIdx.x;

  for (int idx = tid; idx < 128 * 128; idx += 128) {
    int i = idx >> 7, o = idx & 127;
    wt[o * WSTR + i] = (__bf16)Wq[idx];
  }
  for (int idx = tid; idx < 5 * 128; idx += 128)       // zero pad rows 11..15
    ql[(11 + (idx >> 7)) * QSTR + (idx & 127)] = (__bf16)0.f;

  float4 gv = reinterpret_cast<const float4*>(g)[lane];
  float4 bv = reinterpret_cast<const float4*>(bp)[lane];
  for (int r = wave; r < NS_; r += 4) {
    float4 x = reinterpret_cast<const float4*>(qsrc + ((size_t)b * NS_ + r) * C_)[lane];
    float mu = wave_sum32(x.x + x.y + x.z + x.w) * (1.f / 128.f);
    float d0 = x.x - mu, d1 = x.y - mu, d2 = x.z - mu, d3 = x.w - mu;
    float var = wave_sum32(d0*d0 + d1*d1 + d2*d2 + d3*d3) * (1.f / 128.f);
    float rs = rsqrtf(var + LN_EPS_);
    __bf16* dst = &ql[r * QSTR + lane * 4];
    dst[0] = (__bf16)(d0 * rs * gv.x + bv.x);
    dst[1] = (__bf16)(d1 * rs * gv.y + bv.y);
    dst[2] = (__bf16)(d2 * rs * gv.z + bv.z);
    dst[3] = (__bf16)(d3 * rs * gv.w + bv.w);
  }
  __syncthreads();

  const float scale = 0.08838834764831845f;   // 1/sqrt(128)
  const int n = lane & 15, half = (lane >> 4) << 3;
  for (int nt = wave * 2; nt < wave * 2 + 2; ++nt) {
    F8 acc = f8_zero();
#pragma unroll
    for (int kb = 0; kb < 128; kb += 32)
      acc.v = wmma_bf16(load_frag_a(ql, QSTR, kb),
                        load_frag_bt(&wt[nt * 16 * WSTR], WSTR, kb), acc.v);
#pragma unroll
    for (int j = 0; j < 8; ++j) {
      int m = half + j;
      qnp[(size_t)b * 2048 + m * 128 + nt * 16 + n] = (__bf16)(acc.f[j] * scale);
    }
  }
}

// ---------------------------------------------------------------------------
// K3: attention pass. Per (b, key-chunk): logits = qn @ k^T (K tile staged to
// LDS via async-load when available), masked softmax over slots (columns of
// the D-frag, lanes n and n+16 exchange via shfl_xor 16), then U += a @ v via
// WMMA with A = a-tile staged in LDS, B = vT contiguous.  Wave accumulators
// are reduced across the workgroup with LDS atomics, and a per-chunk partial
// U/S is written with plain stores (no global atomics).
// Grid: B*8 blocks of 256 (8 waves, 64 keys/wave, 512 keys/chunk).
// ---------------------------------------------------------------------------
__global__ __launch_bounds__(256) void k_attn(
    const __bf16* __restrict__ k_ws, const __bf16* __restrict__ vT_ws,
    const __bf16* __restrict__ qnp, float* __restrict__ U_part,
    float* __restrict__ S_part)
{
  __shared__ __bf16 albuf[8 * 16 * ASTR];   // per-wave 16x32 a-tiles
#if USE_ASYNC_LDS
  __shared__ __bf16 kstage[8 * 16 * XSTR];  // per-wave staged 16x128 K tiles
#endif
  __shared__ float Ured[16 * C_];
  __shared__ float Sred[16];

  const int tid = threadIdx.x, lane = tid & 31, wave = tid >> 5;
  const int b = blockIdx.x >> 3;
  const int chunk = blockIdx.x & 7;
  const int wkey0 = chunk * 512 + wave * 64;
  __bf16* alds = &albuf[wave * 16 * ASTR];

  for (int i = tid; i < 16 * C_; i += 256) Ured[i] = 0.f;
  if (tid < 16) Sred[tid] = 0.f;
  __syncthreads();

  v16bf qa[4];
#pragma unroll
  for (int kk = 0; kk < 4; ++kk)
    qa[kk] = load_frag_a(qnp + (size_t)b * 2048, 128, kk * 32);

  F8 Uacc[8];
#pragma unroll
  for (int nt = 0; nt < 8; ++nt) Uacc[nt] = f8_zero();
  float Sp[8];
#pragma unroll
  for (int j = 0; j < 8; ++j) Sp[j] = 0.f;

  const int n = lane & 15, half = (lane >> 4) << 3;
  const int nvalid = (lane < 16) ? 8 : 3;   // slots m<11 only

  for (int gi = 0; gi < 2; ++gi) {          // two 32-key groups
    const int kg = wkey0 + gi * 32;
#pragma unroll
    for (int s = 0; s < 2; ++s) {           // two 16-key subtiles
      const __bf16* kb_base = k_ws + ((size_t)b * NK_ + kg + s * 16) * C_;
      F8 L = f8_zero();
#if USE_ASYNC_LDS
      // async-stage 16 keys x 128 chans (4 KB) into wave-private LDS
      __bf16* kst = &kstage[wave * 16 * XSTR];
#pragma unroll
      for (int i = 0; i < 8; ++i) {
        int cidx = i * 32 + lane;           // 256 x b128 chunks
        int row = cidx >> 4, col = cidx & 15;
        async_ld_b128(kb_base + row * C_ + col * 8, kst + row * XSTR + col * 8);
      }
      __builtin_amdgcn_s_wait_asynccnt(0);
#pragma unroll
      for (int kk = 0; kk < 4; ++kk)
        L.v = wmma_bf16(qa[kk], load_frag_bt(kst, XSTR, kk * 32), L.v);
#else
      __builtin_prefetch(kb_base + 32 * C_, 0, 1);
#pragma unroll
      for (int kk = 0; kk < 4; ++kk)
        L.v = wmma_bf16(qa[kk], load_frag_bt(kb_base, C_, kk * 32), L.v);
#endif
      // masked column softmax over slot axis
      float mx = -1e30f;
#pragma unroll
      for (int j = 0; j < 8; ++j) if (j < nvalid) mx = fmaxf(mx, L.f[j]);
      mx = fmaxf(mx, __shfl_xor(mx, 16, 32));
      float e[8], ssum = 0.f;
#pragma unroll
      for (int j = 0; j < 8; ++j) {
        e[j] = (j < nvalid) ? __expf(L.f[j] - mx) : 0.f;
        ssum += e[j];
      }
      float inv = 1.f / (ssum + __shfl_xor(ssum, 16, 32));
#pragma unroll
      for (int j = 0; j < 8; ++j) {
        float a = e[j] * inv;
        Sp[j] += a;
        alds[(j + half) * ASTR + s * 16 + n] = (__bf16)a;
      }
    }
    // U += a(16x32) @ v(32x16) per channel tile
    v16bf af = load_frag_a(alds, ASTR, 0);
#pragma unroll
    for (int nt = 0; nt < 8; ++nt) {
      const __bf16* vt_base = vT_ws + ((size_t)b * C_ + nt * 16) * NK_ + kg;
      Uacc[nt].v = wmma_bf16(af, load_frag_bt(vt_base, NK_, 0), Uacc[nt].v);
    }
  }

  // reduce S partials across the 16 lanes of each half
#pragma unroll
  for (int j = 0; j < 8; ++j) {
    float s = Sp[j];
#pragma unroll
    for (int msk = 1; msk < 16; msk <<= 1) s += __shfl_xor(s, msk, 32);
    Sp[j] = s;
  }
  // cross-wave reduction in LDS (ds_add_f32), then plain partial stores
  if (n == 0) {
#pragma unroll
    for (int j = 0; j < 8; ++j) atomicAdd(&Sred[half + j], Sp[j]);
  }
#pragma unroll
  for (int nt = 0; nt < 8; ++nt)
#pragma unroll
    for (int j = 0; j < 8; ++j)
      atomicAdd(&Ured[(half + j) * C_ + nt * 16 + n], Uacc[nt].f[j]);
  __syncthreads();

  float* up = U_part + (size_t)(b * 8 + chunk) * (16 * C_);
  for (int i = tid; i < 16 * C_; i += 256) up[i] = Ured[i];
  if (tid < 16) S_part[(b * 8 + chunk) * 16 + tid] = Sred[tid];
}

// ---------------------------------------------------------------------------
// K4: upd = (sum_chunks U)/(sum_chunks S + eps); GRU cell via 6 WMMA GEMMs
// (weights phased through one transposed LDS buffer); writes new query.
// Grid: B blocks of 128.
// ---------------------------------------------------------------------------
__global__ __launch_bounds__(128) void k_gru(
    const float* __restrict__ qsrc, const float* __restrict__ U_part,
    const float* __restrict__ S_part,
    const float* __restrict__ Wir, const float* __restrict__ bir,
    const float* __restrict__ Wiz, const float* __restrict__ biz,
    const float* __restrict__ Whr, const float* __restrict__ Whz,
    const float* __restrict__ Win, const float* __restrict__ bin_,
    const float* __restrict__ Whn, const float* __restrict__ bhn,
    float* __restrict__ out)
{
  __shared__ __bf16 ql[16 * QSTR];
  __shared__ __bf16 ul[16 * QSTR];
  __shared__ __bf16 wt[128 * WSTR];
  const int tid = threadIdx.x, lane = tid & 31, wave = tid >> 5;
  const int b = blockIdx.x;

  for (int idx = tid; idx < 16 * 128; idx += 128) {
    int m = idx >> 7, c = idx & 127;
    float qv = 0.f, uv = 0.f;
    if (m < NS_) {
      qv = qsrc[((size_t)b * NS_ + m) * C_ + c];
      float sv = 0.f;
#pragma unroll
      for (int cc = 0; cc < 8; ++cc) {
        uv += U_part[((size_t)(b * 8 + cc) * 16 + m) * C_ + c];
        sv += S_part[(b * 8 + cc) * 16 + m];
      }
      uv /= (sv + ATTN_EPS_);
    }
    ql[m * QSTR + c] = (__bf16)qv;
    ul[m * QSTR + c] = (__bf16)uv;
  }

  const int n = lane & 15, half = (lane >> 4) << 3;
  const int nt0 = wave * 2;

  F8 acc[4][2];   // R, Z, In(=upd@Win+b_in), Hn(=q@Whn+b_hn) x 2 chan tiles
  const float* biases[4] = {bir, biz, bin_, bhn};
#pragma unroll
  for (int a = 0; a < 4; ++a)
#pragma unroll
    for (int t = 0; t < 2; ++t) {
      float bv = biases[a][(nt0 + t) * 16 + n];
#pragma unroll
      for (int j = 0; j < 8; ++j) acc[a][t].f[j] = bv;
    }

  const float* Ws[6]   = {Wir, Whr, Wiz, Whz, Win, Whn};
  const int accIdx[6]  = {0, 0, 1, 1, 2, 3};
  const bool useU[6]   = {true, false, true, false, true, false};
#pragma unroll
  for (int w = 0; w < 6; ++w) {
    __syncthreads();                       // prior consumers done / ql,ul ready
    for (int idx = tid; idx < 128 * 128; idx += 128) {
      int i = idx >> 7, o = idx & 127;
      wt[o * WSTR + i] = (__bf16)Ws[w][idx];
    }
    __syncthreads();
    const __bf16* A = useU[w] ? ul : ql;
    const int ai = accIdx[w];
#pragma unroll
    for (int t = 0; t < 2; ++t)
#pragma unroll
      for (int kb = 0; kb < 128; kb += 32)
        acc[ai][t].v = wmma_bf16(load_frag_a(A, QSTR, kb),
                                 load_frag_bt(&wt[(nt0 + t) * 16 * WSTR], WSTR, kb),
                                 acc[ai][t].v);
  }

#pragma unroll
  for (int t = 0; t < 2; ++t) {
    int c = (nt0 + t) * 16 + n;
#pragma unroll
    for (int j = 0; j < 8; ++j) {
      int m = half + j;
      if (m < NS_) {
        float r  = 1.f / (1.f + __expf(-acc[0][t].f[j]));
        float z  = 1.f / (1.f + __expf(-acc[1][t].f[j]));
        float nn = tanhf(acc[2][t].f[j] + r * acc[3][t].f[j]);
        float qv = qsrc[((size_t)b * NS_ + m) * C_ + c];
        out[((size_t)b * NS_ + m) * C_ + c] = (1.f - z) * nn + z * qv;
      }
    }
  }
}

// ---------------------------------------------------------------------------
extern "C" void kernel_launch(void* const* d_in, const int* in_sizes, int n_in,
                              void* d_out, int out_size, void* d_ws, size_t ws_size,
                              hipStream_t stream) {
  (void)in_sizes; (void)n_in; (void)out_size; (void)ws_size;
  const float* query   = (const float*)d_in[0];
  const float* inputs  = (const float*)d_in[1];
  const float* ln_q_g  = (const float*)d_in[2];
  const float* ln_q_b  = (const float*)d_in[3];
  const float* ln_kv_g = (const float*)d_in[4];
  const float* ln_kv_b = (const float*)d_in[5];
  const float* Wq      = (const float*)d_in[6];
  const float* Wk      = (const float*)d_in[7];
  const float* Wv      = (const float*)d_in[8];
  const float* Wir     = (const float*)d_in[9];
  const float* bir     = (const float*)d_in[10];
  const float* Wiz     = (const float*)d_in[11];
  const float* biz     = (const float*)d_in[12];
  const float* Whr     = (const float*)d_in[13];
  const float* Whz     = (const float*)d_in[14];
  const float* Win     = (const float*)d_in[15];
  const float* bin_    = (const float*)d_in[16];
  const float* Whn     = (const float*)d_in[17];
  const float* bhn     = (const float*)d_in[18];
  float* out = (float*)d_out;

  // workspace layout (~133 MB): k bf16 | vT bf16 | qnp bf16 | U_part | S_part
  char* ws = (char*)d_ws;
  __bf16* k_ws   = (__bf16*)ws;
  __bf16* vT_ws  = k_ws + (size_t)B_ * NK_ * C_;
  __bf16* qnp    = vT_ws + (size_t)B_ * C_ * NK_;
  float*  U_part = (float*)(qnp + (size_t)B_ * 16 * C_);   // [B][8][16][128]
  float*  S_part = U_part + (size_t)B_ * 8 * 16 * C_;      // [B][8][16]

  k_lnkv<<<B_ * (NK_ / 64), 256, 0, stream>>>(inputs, ln_kv_g, ln_kv_b, Wk, Wv,
                                              k_ws, vT_ws);
  for (int it = 0; it < 2; ++it) {
    const float* qs = (it == 0) ? query : out;
    k_qn<<<B_, 128, 0, stream>>>(qs, ln_q_g, ln_q_b, Wq, qnp);
    k_attn<<<B_ * 8, 256, 0, stream>>>(k_ws, vT_ws, qnp, U_part, S_part);
    k_gru<<<B_, 128, 0, stream>>>(qs, U_part, S_part, Wir, bir, Wiz, biz, Whr,
                                  Whz, Win, bin_, Whn, bhn, out);
  }
}
